// BiLSTM_CRF_38001870635614
// MI455X (gfx1250) — matrix operand order repaired and checked by reading, not compiled, not run on hardware
//
#include <hip/hip_runtime.h>
#include <hip/hip_bf16.h>

typedef __bf16 bf16_t;
typedef __attribute__((ext_vector_type(16))) __bf16 v16bf;
typedef __attribute__((ext_vector_type(8)))  float  v8f;

#define GAS __attribute__((address_space(1)))

#define Bq 64
#define Sq 512
#define Eq 128
#define Hq 256
#define Cq 9
#define Gq 1024   /* 4*H */
#define BSq (Bq*Sq)

__device__ __forceinline__ float sigf(float x) { return 1.0f / (1.0f + __expf(-x)); }

// ---------------- embedding gather (fp32 -> bf16) ----------------
__global__ void embed_kernel(const int* __restrict__ ids, const float* __restrict__ emb,
                             bf16_t* __restrict__ x) {
  int bs = blockIdx.x;
  int tok = ids[bs];
  x[(size_t)bs * Eq + threadIdx.x] = (bf16_t)emb[(size_t)tok * Eq + threadIdx.x];
}

// ---------------- pack [w_ih | w_hh] into WMMA B-fragment order ----------------
// fragment index = nt*KB + kb ; element = lane*16 + j
// B layout (32x16 bf16): n = nt*16 + (lane&15) ; k = kb*32 + (j>>3)*16 + (lane>>4)*8 + (j&7)
__global__ void pack_wcat_kernel(const float* __restrict__ w_ih, const float* __restrict__ w_hh,
                                 int Kx, int KB, bf16_t* __restrict__ out) {
  int frag = blockIdx.x;
  int kb = frag % KB;
  int nt = frag / KB;
  int lane = threadIdx.x >> 4;   // 0..31
  int j    = threadIdx.x & 15;   // 0..15
  int n = nt * 16 + (lane & 15);
  int k = kb * 32 + ((j >> 3) << 4) + ((lane >> 4) << 3) + (j & 7);
  float v = (k < Kx) ? w_ih[(size_t)n * Kx + k] : w_hh[(size_t)n * Hq + (k - Kx)];
  out[(size_t)frag * 512 + threadIdx.x] = (bf16_t)v;
}

__global__ void add_bias_kernel(const float* __restrict__ a, const float* __restrict__ b,
                                float* __restrict__ out) {
  int i = blockIdx.x * blockDim.x + threadIdx.x;
  if (i < Gq) out[i] = a[i] + b[i];
}

// ---------------- one LSTM direction: whole time scan in one block ----------------
union AFrag { v16bf v; unsigned int u[8]; };

template <int KX>
__global__ __launch_bounds__(1024, 1)
void lstm_dir_kernel(const bf16_t* __restrict__ x,
                     const bf16_t* __restrict__ wcat_base,
                     const float* __restrict__ bias_base,
                     bf16_t* __restrict__ hout) {
  constexpr int Kin = KX + Hq;
  constexpr int KB  = Kin >> 5;

  extern __shared__ char smem[];
  bf16_t* h_lds = (bf16_t*)smem;                                     // B x H   (32 KB)
  float*  gates = (float*)(smem + (size_t)Bq * Hq * sizeof(bf16_t)); // B x 4H (256 KB)

  const int dir  = blockIdx.x;
  const int rev  = dir;                 // dir 1 runs reversed
  const bf16_t* w    = wcat_base + (size_t)dir * Gq * Kin;
  const float*  bvec = bias_base + dir * Gq;
  const int out_off  = dir * Hq;

  const int tid  = threadIdx.x;
  const int lane = tid & 31;
  const int wave = tid >> 5;
  const int mt   = wave & 3;            // batch tile 0..3  (64 rows)
  const int ntb  = (wave >> 2) << 3;    // 8 gate tiles per wave (64 total)
  const int half = lane >> 4;
  const int l15  = lane & 15;
  const int bA   = mt * 16 + l15;       // A-matrix row (batch) for this lane

  // Per-wave fragment base: all fragment addresses become base + compile-time imm.
  const bf16_t* wwave = w + (size_t)ntb * KB * 512 + lane * 16;

  // zero recurrent state
  for (int r = 0; r < (Bq * Hq) / 1024; ++r) h_lds[tid + r * 1024] = (bf16_t)0.0f;
  float c[16];
#pragma unroll
  for (int r = 0; r < 16; ++r) c[r] = 0.0f;
  __syncthreads();

  const int bc = tid >> 4;              // cell-update batch row
  const int ub = (tid & 15) << 4;       // cell-update unit base

  for (int t = 0; t < Sq; ++t) {
    const int s = rev ? (Sq - 1 - t) : t;

    // Opaque copy of the weight base pointer in ADDRESS SPACE 1 (global):
    // - blocks LICM from hoisting the loop-invariant weight-fragment loads out
    //   of the 512-step scan (would spill ~48-96 KB/wave to scratch);
    // - keeps the global address space so loads lower to global_load_b128
    //   (LOADcnt only), not flat_load (LOADcnt+DScnt, LDS-path contention).
    const GAS bf16_t* wt = (const GAS bf16_t*)wwave;
    asm volatile("" : "+v"(wt));

    // ---- gates = [x_t | h_prev] @ Wcat^T + bias, 64x1024 via WMMA tiles ----
    v8f acc[8];
    const v8f vzero = {0.f, 0.f, 0.f, 0.f, 0.f, 0.f, 0.f, 0.f};
#pragma unroll
    for (int i = 0; i < 8; ++i) acc[i] = vzero;

    const bf16_t* xrow = x + ((size_t)bA * Sq + s) * KX;
    const bf16_t* hrow = h_lds + bA * Hq;

#pragma unroll
    for (int kb = 0; kb < KB; ++kb) {
      const int kbase = kb << 5;        // compile-time after unroll
      AFrag a;
      if (kbase < KX) {                 // resolved at compile time per iteration
#pragma unroll
        for (int v = 0; v < 8; ++v) {
          int k = kbase + ((v >> 2) << 4) + (half << 3) + ((v & 3) << 1);
          a.u[v] = *(const unsigned int*)(xrow + k);
        }
      } else {
#pragma unroll
        for (int v = 0; v < 8; ++v) {
          int k = (kbase - KX) + ((v >> 2) << 4) + (half << 3) + ((v & 3) << 1);
          a.u[v] = *(const unsigned int*)(hrow + k);
        }
      }
      // B fragments in batches of 4: one load clause -> 4 back-to-back WMMAs
#pragma unroll
      for (int g = 0; g < 2; ++g) {
        v16bf bf[4];
#pragma unroll
        for (int q = 0; q < 4; ++q)
          bf[q] = *(const GAS v16bf*)(wt + (size_t)((g * 4 + q) * KB + kb) * 512);
#pragma unroll
        for (int q = 0; q < 4; ++q)
          acc[g * 4 + q] = __builtin_amdgcn_wmma_f32_16x16x32_bf16(
              false, a.v, false, bf[q], (short)0, acc[g * 4 + q], false, false);
      }
    }

#pragma unroll
    for (int i = 0; i < 8; ++i) {
      int nt = ntb + i;
      float bb = bvec[nt * 16 + l15];
#pragma unroll
      for (int j = 0; j < 8; ++j) {     // D layout: m = j + 8*half, n = lane&15
        int brow = mt * 16 + (half << 3) + j;
        gates[brow * Gq + nt * 16 + l15] = acc[i][j] + bb;
      }
    }
    __syncthreads();

    // ---- LSTM cell: c kept in registers across the whole scan ----
#pragma unroll
    for (int r = 0; r < 16; ++r) {
      int u = ub + r;
      float ig = gates[bc * Gq + u];
      float fg = gates[bc * Gq + Hq + u];
      float gg = gates[bc * Gq + 2 * Hq + u];
      float og = gates[bc * Gq + 3 * Hq + u];
      float cc = sigf(fg) * c[r] + sigf(ig) * tanhf(gg);
      c[r] = cc;
      float hh = sigf(og) * tanhf(cc);
      h_lds[bc * Hq + u] = (bf16_t)hh;
      hout[((size_t)bc * Sq + s) * (2 * Hq) + out_off + u] = (bf16_t)hh;
    }
    __syncthreads();
  }
}

// ---------------- classifier: emissions = h1 @ cls_w.T + cls_b ----------------
__global__ void classifier_kernel(const bf16_t* __restrict__ h, const float* __restrict__ w,
                                  const float* __restrict__ bb, float* __restrict__ em) {
  int idx = blockIdx.x * blockDim.x + threadIdx.x;
  if (idx >= BSq * Cq) return;
  int bs = idx / Cq, cc = idx % Cq;
  const bf16_t* hr = h + (size_t)bs * (2 * Hq);
  const float*  wr = w + (size_t)cc * (2 * Hq);
  float s = bb[cc];
  for (int k = 0; k < 2 * Hq; ++k) s += (float)hr[k] * wr[k];
  em[(size_t)bs * Cq + cc] = s;
}

// ---------------- CRF log-likelihood (one thread per batch, C=9) ----------------
__global__ void crf_llh_kernel(const float* __restrict__ em, const int* __restrict__ tags,
                               const int* __restrict__ mask,
                               const float* __restrict__ startv, const float* __restrict__ endv,
                               const float* __restrict__ trans, float* __restrict__ llh) {
  int b = threadIdx.x;
  if (b >= Bq) return;
  const float* e = em + (size_t)b * Sq * Cq;
  const int* tg = tags + (size_t)b * Sq;
  const int* mk = mask + (size_t)b * Sq;

  int t0 = tg[0];
  float num = startv[t0] + e[t0];
  int msum = mk[0];
  for (int t = 1; t < Sq; ++t) {
    int tt = tg[t];
    float m = (float)mk[t];
    num += m * (trans[tg[t - 1] * Cq + tt] + e[t * Cq + tt]);
    msum += mk[t];
  }
  num += endv[tg[msum - 1]];

  float alpha[Cq];
  for (int j = 0; j < Cq; ++j) alpha[j] = startv[j] + e[j];
  for (int t = 1; t < Sq; ++t) {
    float na[Cq];
    for (int j = 0; j < Cq; ++j) {
      float mx = -1e30f;
      for (int i = 0; i < Cq; ++i) mx = fmaxf(mx, alpha[i] + trans[i * Cq + j]);
      float sm = 0.f;
      for (int i = 0; i < Cq; ++i) sm += __expf(alpha[i] + trans[i * Cq + j] - mx);
      na[j] = mx + __logf(sm) + e[t * Cq + j];
    }
    bool on = mk[t] > 0;
    for (int j = 0; j < Cq; ++j) alpha[j] = on ? na[j] : alpha[j];
  }
  float mx = -1e30f;
  for (int j = 0; j < Cq; ++j) mx = fmaxf(mx, alpha[j] + endv[j]);
  float sm = 0.f;
  for (int j = 0; j < Cq; ++j) sm += __expf(alpha[j] + endv[j] - mx);
  llh[b] = num - (mx + __logf(sm));
}

// ---------------- Viterbi decode + loss reduction ----------------
__global__ void viterbi_loss_kernel(const float* __restrict__ em, const int* __restrict__ mask,
                                    const float* __restrict__ startv, const float* __restrict__ endv,
                                    const float* __restrict__ trans, const float* __restrict__ llh,
                                    int* __restrict__ bp, float* __restrict__ out) {
  int b = threadIdx.x;
  if (b == 0) {
    float s = 0.f;
    for (int i = 0; i < Bq; ++i) s += llh[i];
    out[0] = -s / (float)Bq;
  }
  if (b >= Bq) return;
  const float* e = em + (size_t)b * Sq * Cq;
  const int* mk = mask + (size_t)b * Sq;
  int* bpb = bp + (size_t)b * Sq * Cq;

  float sc[Cq];
  for (int j = 0; j < Cq; ++j) sc[j] = startv[j] + e[j];
  for (int t = 1; t < Sq; ++t) {
    float ns[Cq]; int nb[Cq];
    for (int j = 0; j < Cq; ++j) {
      float best = sc[0] + trans[j];
      int bi = 0;
      for (int i = 1; i < Cq; ++i) {
        float v = sc[i] + trans[i * Cq + j];
        if (v > best) { best = v; bi = i; }
      }
      ns[j] = best + e[t * Cq + j];
      nb[j] = bi;
    }
    bool on = mk[t] > 0;
    for (int j = 0; j < Cq; ++j) {
      sc[j] = on ? ns[j] : sc[j];
      bpb[t * Cq + j] = on ? nb[j] : j;
    }
  }
  float best = sc[0] + endv[0];
  int cur = 0;
  for (int j = 1; j < Cq; ++j) {
    float v = sc[j] + endv[j];
    if (v > best) { best = v; cur = j; }
  }
  out[1 + (size_t)b * Sq + (Sq - 1)] = (float)cur;
  for (int t = Sq - 1; t >= 1; --t) {
    cur = bpb[t * Cq + cur];
    out[1 + (size_t)b * Sq + (t - 1)] = (float)cur;
  }
}

extern "C" void kernel_launch(void* const* d_in, const int* in_sizes, int n_in,
                              void* d_out, int out_size, void* d_ws, size_t ws_size,
                              hipStream_t stream) {
  (void)in_sizes; (void)n_in; (void)out_size; (void)ws_size;
  const int*   token_ids = (const int*)d_in[0];
  const int*   labels    = (const int*)d_in[1];
  const int*   amask     = (const int*)d_in[2];
  const float* emb       = (const float*)d_in[3];
  const float* w_ih0     = (const float*)d_in[4];
  const float* w_hh0     = (const float*)d_in[5];
  const float* b_ih0     = (const float*)d_in[6];
  const float* b_hh0     = (const float*)d_in[7];
  const float* w_ih1     = (const float*)d_in[8];
  const float* w_hh1     = (const float*)d_in[9];
  const float* b_ih1     = (const float*)d_in[10];
  const float* b_hh1     = (const float*)d_in[11];
  const float* cls_w     = (const float*)d_in[12];
  const float* cls_b     = (const float*)d_in[13];
  const float* startv    = (const float*)d_in[14];
  const float* endv      = (const float*)d_in[15];
  const float* trans     = (const float*)d_in[16];
  float* out = (float*)d_out;

  // workspace sub-allocation (~80 MB total)
  char* ws = (char*)d_ws;
  size_t off = 0;
  auto alloc = [&](size_t bytes) -> char* {
    off = (off + 255) & ~(size_t)255;
    char* p = ws + off;
    off += bytes;
    return p;
  };
  bf16_t* x0    = (bf16_t*)alloc((size_t)BSq * Eq * 2);          //  8 MB
  bf16_t* h0    = (bf16_t*)alloc((size_t)BSq * 2 * Hq * 2);      // 32 MB
  bf16_t* h1    = (bf16_t*)alloc((size_t)BSq * 2 * Hq * 2);      // 32 MB
  bf16_t* wc0   = (bf16_t*)alloc((size_t)2 * Gq * (Eq + Hq) * 2);
  bf16_t* wc1   = (bf16_t*)alloc((size_t)2 * Gq * (3 * Hq) * 2);
  float*  bias0 = (float*)alloc((size_t)2 * Gq * 4);
  float*  bias1 = (float*)alloc((size_t)2 * Gq * 4);
  float*  emis  = (float*)alloc((size_t)BSq * Cq * 4);
  float*  llh   = (float*)alloc((size_t)Bq * 4);
  int*    bp    = (int*)alloc((size_t)Bq * Sq * Cq * 4);

  const size_t lds_bytes = (size_t)Bq * Hq * 2 + (size_t)Bq * Gq * 4; // 288 KB
  hipFuncSetAttribute((const void*)lstm_dir_kernel<Eq>,
                      hipFuncAttributeMaxDynamicSharedMemorySize, (int)lds_bytes);
  hipFuncSetAttribute((const void*)lstm_dir_kernel<2 * Hq>,
                      hipFuncAttributeMaxDynamicSharedMemorySize, (int)lds_bytes);

  embed_kernel<<<BSq, Eq, 0, stream>>>(token_ids, emb, x0);

  for (int d = 0; d < 2; ++d) {
    pack_wcat_kernel<<<64 * 12, 512, 0, stream>>>(w_ih0 + (size_t)d * Gq * Eq,
                                                  w_hh0 + (size_t)d * Gq * Hq,
                                                  Eq, 12, wc0 + (size_t)d * Gq * (Eq + Hq));
    pack_wcat_kernel<<<64 * 24, 512, 0, stream>>>(w_ih1 + (size_t)d * Gq * 2 * Hq,
                                                  w_hh1 + (size_t)d * Gq * Hq,
                                                  2 * Hq, 24, wc1 + (size_t)d * Gq * (3 * Hq));
    add_bias_kernel<<<4, 256, 0, stream>>>(b_ih0 + d * Gq, b_hh0 + d * Gq, bias0 + d * Gq);
    add_bias_kernel<<<4, 256, 0, stream>>>(b_ih1 + d * Gq, b_hh1 + d * Gq, bias1 + d * Gq);
  }

  lstm_dir_kernel<Eq><<<2, 1024, lds_bytes, stream>>>(x0, wc0, bias0, h0);
  lstm_dir_kernel<2 * Hq><<<2, 1024, lds_bytes, stream>>>(h0, wc1, bias1, h1);

  classifier_kernel<<<(BSq * Cq + 255) / 256, 256, 0, stream>>>(h1, cls_w, cls_b, emis);
  crf_llh_kernel<<<1, 64, 0, stream>>>(emis, labels, amask, startv, endv, trans, llh);
  viterbi_loss_kernel<<<1, 64, 0, stream>>>(emis, amask, startv, endv, trans, llh, bp, out);
}
